// JointWrapper_66709432041823
// MI455X (gfx1250) — matrix operand order repaired and checked by reading, not compiled, not run on hardware
//
#include <hip/hip_runtime.h>

// ---------------------------------------------------------------------------
// RNN-T joint network for MI455X (gfx1250, wave32, WMMA).
//   enc_proj[b,t,h] = sum_d enc[b,d,t]*W_enc[d,h] + b_enc[h]      (f32->f16, tiny)
//   dec_proj[b,u,h] = sum_d dec[b,d,u]*W_pred[d,h] + b_pred[h]    (f32->f16, tiny)
//   out[b,t,u,v]    = relu(enc_proj+dec_proj) @ W_out + b_out     (f16 WMMA, f32 acc)
// Joint GEMM: M=80000, K=640, N=1025 (padded 1152). 105 GFLOP, AI~320 F/B
// vs output stream (328 MB) -> compute bound -> f16 WMMA path, NT stores.
// ---------------------------------------------------------------------------

typedef __attribute__((ext_vector_type(16))) _Float16 v16h;
typedef __attribute__((ext_vector_type(8)))  _Float16 v8h;
typedef __attribute__((ext_vector_type(8)))  float    v8f;

#define B_    4
#define T_    200
#define U_    100
#define DENC_ 512
#define DDEC_ 640
#define H_    640
#define V_    1025
#define VP_   1152             // V padded to 9*128 (N-tile 128), zero tail
#define MTOT_ (B_ * T_ * U_)   // 80000, exact multiple of 64

#define LDA_  40               // LDS A row stride (halfs): 80B, 16B-aligned, no 4-way conflicts
#define LDB_  136              // LDS B row stride (halfs): 272B, 16B-aligned, <=2-way conflicts

// ---------------------------------------------------------------------------
// Kernel 0: W_out f32[H][V] -> f16[H][VP], zero-padded columns >= V
// ---------------------------------------------------------------------------
__global__ void wout_to_f16(const float* __restrict__ W, _Float16* __restrict__ Wh) {
    int idx = blockIdx.x * 256 + threadIdx.x;
    if (idx >= H_ * VP_) return;
    int r = idx / VP_;
    int c = idx - r * VP_;
    float v = (c < V_) ? W[(size_t)r * V_ + c] : 0.0f;
    Wh[idx] = (_Float16)v;
}

// ---------------------------------------------------------------------------
// Kernels 1/2: small input projections (<1% of FLOPs), f32 math, f16 output.
// X is [B, D, L] (d-major, l-minor); out is [B, L, H] row-major f16.
// ---------------------------------------------------------------------------
__global__ void proj_kernel(const float* __restrict__ X, const float* __restrict__ W,
                            const float* __restrict__ bias, _Float16* __restrict__ out,
                            int D, int L) {
    int idx = blockIdx.x * blockDim.x + threadIdx.x;
    int total = B_ * L * H_;
    if (idx >= total) return;
    int h = idx % H_;
    int l = (idx / H_) % L;
    int b = idx / (H_ * L);
    const float* x = X + (size_t)b * D * L + l;   // stride L over d
    const float* w = W + h;                        // stride H over d
    float acc = bias[h];
#pragma unroll 4
    for (int d = 0; d < D; ++d)
        acc = fmaf(x[(size_t)d * L], w[(size_t)d * H_], acc);
    out[idx] = (_Float16)acc;
}

// ---------------------------------------------------------------------------
// Kernel 3: joint GEMM.  Block: 256 threads = 8 waves, C tile 64(M) x 128(N).
// Wave grid 2x4; each wave owns a 32x32 C tile = 4 accumulators fed by
// 2 A frags x 2 B frags -> 4 v_wmma_f32_16x16x32_f16 per 32-wide K step.
// ---------------------------------------------------------------------------
__global__ __launch_bounds__(256) void joint_gemm(
        const _Float16* __restrict__ enc_proj,   // [B*T, H] f16
        const _Float16* __restrict__ dec_proj,   // [B*U, H] f16
        const _Float16* __restrict__ Wh,         // [H, VP] f16
        const float* __restrict__ b_out,         // [V]
        float* __restrict__ out)                 // [MTOT, V]
{
    __shared__ _Float16 ldsA[64 * LDA_];   // A tile rows 0..63, k 0..31
    __shared__ _Float16 ldsB[32 * LDB_];   // B tile k 0..31, n 0..127

    const int tid  = threadIdx.x;
    const int lane = tid & 31;
    const int wv   = tid >> 5;
    const int m0   = blockIdx.y * 64;
    const int n0   = blockIdx.x * 128;

    const int mw = (wv >> 2) * 32;    // wave M offset within block tile
    const int nw = (wv & 3) * 32;     // wave N offset within block tile

    // --- cooperative A-tile loader: 1 row x 8 cols per thread ---------------
    const int rA = tid >> 2;          // 0..63
    const int cA = (tid & 3) * 8;     // 0,8,16,24
    {
        // row index -> (b,t,u); MTOT multiple of 64 -> always valid
    }
    const int mRow = m0 + rA;
    const int bIdx = mRow / (T_ * U_);
    const int rem  = mRow - bIdx * (T_ * U_);
    const int tIdx = rem / U_;
    const int uIdx = rem - tIdx * U_;
    const _Float16* erow = enc_proj + (size_t)(bIdx * T_ + tIdx) * H_ + cA;
    const _Float16* drow = dec_proj + (size_t)(bIdx * U_ + uIdx) * H_ + cA;
    _Float16* aDst = &ldsA[rA * LDA_ + cA];

    // --- cooperative B-tile loader: 1 k-row x 16 cols per thread ------------
    const int rB = tid >> 3;          // 0..31
    const int cB = (tid & 7) * 16;    // 0..112
    const _Float16* wsrc = Wh + (size_t)rB * VP_ + n0 + cB;
    _Float16* bDst = &ldsB[rB * LDB_ + cB];

    // --- fragment addresses (ISA 7.12.2 layouts) ----------------------------
    // A (16-bit 16x32): lanes 0-15: row=lane, K 0..7 & 16..23; lanes 16-31:
    // same rows, K 8..15 & 24..31.
    // B: lane%16 = N column; K 0..7 & 16..23 (lanes 0-15), K 8..15 & 24..31 (lanes 16-31).
    const int fr  = lane & 15;
    const int sel = lane >> 4;
    const _Float16* aF0 = &ldsA[(mw + fr) * LDA_ + sel * 8];
    const _Float16* aF1 = aF0 + 16 * LDA_;
    const _Float16* bF0 = &ldsB[(sel * 8) * LDB_ + nw + fr];
    const _Float16* bF1 = bF0 + 16;

    v8f c00 = {}, c01 = {}, c10 = {}, c11 = {};

    for (int k0 = 0; k0 < H_; k0 += 32) {
        __syncthreads();   // previous iteration's fragment reads complete

        // A tile: relu(enc_row + dec_row) in packed f16
        {
            v8h e = *(const v8h*)(erow + k0);
            v8h d = *(const v8h*)(drow + k0);
            v8h s = e + d;
#pragma unroll
            for (int i = 0; i < 8; ++i)
                s[i] = (s[i] > (_Float16)0) ? s[i] : (_Float16)0;
            *(v8h*)aDst = s;
        }
        // B tile: two 16B-aligned global loads (VP padding guarantees bounds)
        {
            const _Float16* wp = wsrc + (size_t)k0 * VP_;
            *(v8h*)bDst       = *(const v8h*)wp;
            *(v8h*)(bDst + 8) = *(const v8h*)(wp + 8);
        }
        __syncthreads();

        union Frag { v16h v; v8h h[2]; };
        Frag a0, a1, bb0, bb1;
        a0.h[0]  = *(const v8h*)(aF0);
        a0.h[1]  = *(const v8h*)(aF0 + 16);
        a1.h[0]  = *(const v8h*)(aF1);
        a1.h[1]  = *(const v8h*)(aF1 + 16);
#pragma unroll
        for (int i = 0; i < 8; ++i) {
            bb0.v[i]     = bF0[i * LDB_];
            bb0.v[8 + i] = bF0[(16 + i) * LDB_];
            bb1.v[i]     = bF1[i * LDB_];
            bb1.v[8 + i] = bF1[(16 + i) * LDB_];
        }

        c00 = __builtin_amdgcn_wmma_f32_16x16x32_f16(
            false, a0.v, false, bb0.v, (short)0, c00, false, false);
        c01 = __builtin_amdgcn_wmma_f32_16x16x32_f16(
            false, a0.v, false, bb1.v, (short)0, c01, false, false);
        c10 = __builtin_amdgcn_wmma_f32_16x16x32_f16(
            false, a1.v, false, bb0.v, (short)0, c10, false, false);
        c11 = __builtin_amdgcn_wmma_f32_16x16x32_f16(
            false, a1.v, false, bb1.v, (short)0, c11, false, false);
    }

    // --- store (C/D layout: lane 0-15: N=lane, M=vgpr r; lane 16-31: M+=8) --
    // Non-temporal: 328 MB output stream must not evict L2-resident operands.
    const int nA = n0 + nw + fr;
    const int nB = nA + 16;
    const int mA = m0 + mw + sel * 8;
    const float biasA = (nA < V_) ? b_out[nA] : 0.0f;
    const float biasB = (nB < V_) ? b_out[nB] : 0.0f;
#pragma unroll
    for (int r = 0; r < 8; ++r) {
        size_t ro0 = (size_t)(mA + r) * V_;        // rows mA..mA+7
        size_t ro1 = ro0 + (size_t)16 * V_;        // rows mA+16..mA+23
        if (nA < V_) {
            __builtin_nontemporal_store(c00[r] + biasA, &out[ro0 + nA]);
            __builtin_nontemporal_store(c10[r] + biasA, &out[ro1 + nA]);
        }
        if (nB < V_) {
            __builtin_nontemporal_store(c01[r] + biasB, &out[ro0 + nB]);
            __builtin_nontemporal_store(c11[r] + biasB, &out[ro1 + nB]);
        }
    }
}

// ---------------------------------------------------------------------------
extern "C" void kernel_launch(void* const* d_in, const int* in_sizes, int n_in,
                              void* d_out, int out_size, void* d_ws, size_t ws_size,
                              hipStream_t stream) {
    (void)in_sizes; (void)n_in; (void)out_size; (void)ws_size;

    const float* enc    = (const float*)d_in[0];   // [B, D_ENC, T]
    const float* dec    = (const float*)d_in[1];   // [B, D_DEC, U]
    const float* W_enc  = (const float*)d_in[2];   // [D_ENC, H]
    const float* b_enc  = (const float*)d_in[3];   // [H]
    const float* W_pred = (const float*)d_in[4];   // [D_DEC, H]
    const float* b_pred = (const float*)d_in[5];   // [H]
    const float* W_out  = (const float*)d_in[6];   // [H, V]
    const float* b_out  = (const float*)d_in[7];   // [V]
    float*       out    = (float*)d_out;           // [B, T, U, V]

    // workspace layout (sections are multiples of 256B)
    const size_t encBytes = (size_t)B_ * T_ * H_ * sizeof(_Float16);  // 1,024,000
    const size_t decBytes = (size_t)B_ * U_ * H_ * sizeof(_Float16);  //   512,000
    char* ws = (char*)d_ws;
    _Float16* enc_proj = (_Float16*)ws;
    _Float16* dec_proj = (_Float16*)(ws + encBytes);
    _Float16* Wh       = (_Float16*)(ws + encBytes + decBytes);       // 640*1152 f16

    // K0: weight conversion/padding
    {
        int total = H_ * VP_;
        wout_to_f16<<<(total + 255) / 256, 256, 0, stream>>>(W_out, Wh);
    }
    // K1/K2: projections
    {
        int totalE = B_ * T_ * H_;
        proj_kernel<<<(totalE + 255) / 256, 256, 0, stream>>>(
            enc, W_enc, b_enc, enc_proj, DENC_, T_);
        int totalD = B_ * U_ * H_;
        proj_kernel<<<(totalD + 255) / 256, 256, 0, stream>>>(
            dec, W_pred, b_pred, dec_proj, DDEC_, U_);
    }
    // K3: joint WMMA GEMM — grid: 9 N-tiles x 1250 M-tiles
    {
        dim3 grid(VP_ / 128, MTOT_ / 64);
        joint_gemm<<<grid, 256, 0, stream>>>(enc_proj, dec_proj, Wh, b_out, out);
    }
}
